// EdgeSAGEConv_61134564491909
// MI455X (gfx1250) — matrix operand order repaired and checked by reading, not compiled, not run on hardware
//
#include <hip/hip_runtime.h>

#define H 128
#define LDSK 136   // padded LDS row stride in bf16: 272B = 17*16B aligned, 68 dwords -> no bank conflicts

typedef __attribute__((ext_vector_type(16))) __bf16 v16bf;
typedef __attribute__((ext_vector_type(8)))  float  v8f;
typedef __attribute__((ext_vector_type(4)))  float  v4f;
typedef __attribute__((address_space(3)))    __bf16 lds_bf16;

__device__ __forceinline__ __bf16 f2bf(float f) {
  unsigned u = __builtin_bit_cast(unsigned, f);
  unsigned r = u + 0x7FFFu + ((u >> 16) & 1u);   // round-to-nearest-even
  unsigned short s = (unsigned short)(r >> 16);
  return __builtin_bit_cast(__bf16, s);
}

__device__ __forceinline__ float bf2f(unsigned short s) {
  return __builtin_bit_cast(float, (unsigned)s << 16);
}

// 16 contiguous bf16 (32B, 16B-aligned) -> v16bf  (B fragment: N=lane%16, K=base..base+15)
__device__ __forceinline__ v16bf ld_bf16x16(const __bf16* p) {
  union { v16bf v; v4f h[2]; } u;
  u.h[0] = *(const v4f*)(p);
  u.h[1] = *(const v4f*)(p + 8);
  return u.v;
}

// A fragment, contiguous K order: elems 0..15 <- K=klo..klo+15
__device__ __forceinline__ v16bf ld_afrag(const __bf16* row, int klo) {
  union { v16bf v; v4f h[2]; } u;
  u.h[0] = *(const v4f*)(row + klo);
  u.h[1] = *(const v4f*)(row + klo + 8);
  return u.v;
}

__device__ __forceinline__ v8f wmma_bf16(v16bf a, v16bf b, v8f c) {
  return __builtin_amdgcn_wmma_f32_16x16x32_bf16(false, a, false, b, (short)0, c, false, false);
}

// ---------------- kernel 1: zero accumulators, convert x and weights to bf16 ----------------
__global__ void __launch_bounds__(256)
gnn_init(float* __restrict__ zero_base, size_t zero_count,
         const float* __restrict__ x, __bf16* __restrict__ x_bf, size_t xcount,
         const float* __restrict__ wn, const float* __restrict__ wsf, const float* __restrict__ we,
         __bf16* __restrict__ wn_bf, __bf16* __restrict__ ws_bf, __bf16* __restrict__ we_bf) {
  size_t stride = (size_t)gridDim.x * blockDim.x;
  size_t t0 = (size_t)blockIdx.x * blockDim.x + threadIdx.x;
  for (size_t i = t0; i < zero_count; i += stride) zero_base[i] = 0.0f;
  for (size_t i = t0; i < xcount; i += stride) x_bf[i] = f2bf(x[i]);
  for (size_t i = t0; i < (size_t)H * H; i += stride) {
    wn_bf[i] = f2bf(wn[i]);
    ws_bf[i] = f2bf(wsf[i]);
    we_bf[i] = f2bf(we[i]);
  }
}

// Async-copy a 128x128 bf16 tile (row-major, 256B rows) into padded LDS (272B rows).
__device__ __forceinline__ void stage_tile_async(const __bf16* __restrict__ gsrc,
                                                 int row0, int n, __bf16* lds) {
  lds_bf16* lbase = (lds_bf16*)lds;
  const int tid = threadIdx.x;
#pragma unroll
  for (int it = 0; it < 8; ++it) {              // 2048 16B-chunks / 256 threads
    int i = it * 256 + tid;
    int r = i >> 4;
    int b = (i & 15) << 4;                      // byte offset within a 256B row
    int gr = row0 + r;
    if (gr >= n) gr = n - 1;                    // clamp: stay in-bounds, result discarded
    const char* g = (const char*)gsrc + (size_t)gr * (H * 2) + b;
    lds_bf16* l = lbase + r * LDSK + (b >> 1);
    asm volatile("global_load_async_to_lds_b128 %0, %1, off"
                 :: "v"(l), "v"(g) : "memory");
  }
  asm volatile("s_wait_asynccnt 0x0" ::: "memory");
  __syncthreads();
}

// ---------------- kernel 2: x_nei = x@Wn.T (bf16, ws), x_self = x@Ws.T (f32, d_out) ----------------
__global__ void __launch_bounds__(256)
gnn_node_gemm(const __bf16* __restrict__ x_bf,
              const __bf16* __restrict__ wn_bf,
              const __bf16* __restrict__ ws_bf,
              __bf16* __restrict__ x_nei_bf,
              float* __restrict__ x_self,
              int n) {
  __shared__ __bf16 ldsA[128 * LDSK];
  const int row0 = blockIdx.x * 128;
  stage_tile_async(x_bf, row0, n, ldsA);

  const int tid = threadIdx.x;
  const int wave = tid >> 5, lane = tid & 31;
  const int hi = lane >> 4, lr = lane & 15;
  const bool full = (row0 + 128 <= n);
  const __bf16* arow = &ldsA[(wave * 16 + lr) * LDSK];
  v16bf afrag[4];
#pragma unroll
  for (int kc = 0; kc < 4; ++kc) afrag[kc] = ld_afrag(arow, kc * 32 + hi * 16);

#pragma unroll
  for (int c = 0; c < 8; ++c) {
    v8f accN = {};
    v8f accS = {};
    const int ncol = c * 16 + lr;
    const __bf16* bn = wn_bf + (size_t)ncol * H + hi * 16;
    const __bf16* bs = ws_bf + (size_t)ncol * H + hi * 16;
#pragma unroll
    for (int kc = 0; kc < 4; ++kc) {
      accN = wmma_bf16(afrag[kc], ld_bf16x16(bn + kc * 32), accN);
      accS = wmma_bf16(afrag[kc], ld_bf16x16(bs + kc * 32), accS);
    }
    if (full) {
#pragma unroll
      for (int j = 0; j < 8; ++j) {
        size_t idx = (size_t)(row0 + wave * 16 + j + hi * 8) * H + ncol;
        x_nei_bf[idx] = f2bf(accN[j]);
        x_self[idx]   = accS[j];
      }
    } else {
#pragma unroll
      for (int j = 0; j < 8; ++j) {
        int m = row0 + wave * 16 + j + hi * 8;
        if (m < n) {
          size_t idx = (size_t)m * H + ncol;
          x_nei_bf[idx] = f2bf(accN[j]);
          x_self[idx]   = accS[j];
        }
      }
    }
  }
}

// ---------------- kernel 3: per-edge scatter (linearity: raw edge_attr + gathered bf16 x_nei) ------
__global__ void __launch_bounds__(256)
gnn_edge_scatter(const long long* __restrict__ ei,
                 const float* __restrict__ edge_attr,
                 const __bf16* __restrict__ x_nei_bf,
                 float* __restrict__ acc_n,
                 float* __restrict__ acc_e,
                 float* __restrict__ deg,
                 int ne) {
  int e = blockIdx.x * 8 + (threadIdx.x >> 5);
  if (e >= ne) return;
  int lane = threadIdx.x & 31;
  long long src = ei[e];
  long long dst = ei[(size_t)ne + e];
  v4f ea = *(const v4f*)(edge_attr + (size_t)e * H + lane * 4);
  unsigned long long raw = *(const unsigned long long*)(x_nei_bf + (size_t)src * H + lane * 4);
  float* pe = acc_e + (size_t)dst * H + lane * 4;
  float* pn = acc_n + (size_t)dst * H + lane * 4;
  atomicAdd(pe + 0, ea.x); atomicAdd(pe + 1, ea.y);
  atomicAdd(pe + 2, ea.z); atomicAdd(pe + 3, ea.w);
  atomicAdd(pn + 0, bf2f((unsigned short)raw));
  atomicAdd(pn + 1, bf2f((unsigned short)(raw >> 16)));
  atomicAdd(pn + 2, bf2f((unsigned short)(raw >> 32)));
  atomicAdd(pn + 3, bf2f((unsigned short)(raw >> 48)));
  if (lane == 0) atomicAdd(deg + dst, 1.0f);
}

// ---------------- kernel 4: z = acc_e@We.T; out = relu(x_self + b + (acc_n + z)/max(deg,1)) -------
__global__ void __launch_bounds__(256)
gnn_edge_gemm_epilogue(const float* __restrict__ acc_e,
                       const __bf16* __restrict__ we_bf,
                       const float* __restrict__ acc_n,
                       const float* __restrict__ deg,
                       const float* __restrict__ bias,
                       float* __restrict__ out,   // holds x_self on entry, result on exit
                       int n) {
  __shared__ __bf16 ldsA[128 * LDSK];
  const int tid  = threadIdx.x;
  const int row0 = blockIdx.x * 128;
  // stage fp32 accumulator tile -> bf16 LDS (convert in VGPRs; source is fp32 from atomics)
  for (int i = tid; i < 128 * (H / 4); i += 256) {
    int r = i >> 5;
    int k = (i & 31) << 2;
    v4f v = {0.f, 0.f, 0.f, 0.f};
    int gr = row0 + r;
    if (gr < n) v = *(const v4f*)(acc_e + (size_t)gr * H + k);
    __bf16* d = &ldsA[r * LDSK + k];
    d[0] = f2bf(v.x); d[1] = f2bf(v.y); d[2] = f2bf(v.z); d[3] = f2bf(v.w);
  }
  __syncthreads();

  const int wave = tid >> 5, lane = tid & 31;
  const int hi = lane >> 4, lr = lane & 15;
  const bool full = (row0 + 128 <= n);
  const __bf16* arow = &ldsA[(wave * 16 + lr) * LDSK];
  v16bf afrag[4];
#pragma unroll
  for (int kc = 0; kc < 4; ++kc) afrag[kc] = ld_afrag(arow, kc * 32 + hi * 16);

#pragma unroll
  for (int c = 0; c < 8; ++c) {
    v8f acc = {};
    const int ncol = c * 16 + lr;
    const __bf16* be = we_bf + (size_t)ncol * H + hi * 16;
#pragma unroll
    for (int kc = 0; kc < 4; ++kc)
      acc = wmma_bf16(afrag[kc], ld_bf16x16(be + kc * 32), acc);

    float bn = bias[ncol];
#pragma unroll
    for (int j = 0; j < 8; ++j) {
      int m = row0 + wave * 16 + j + hi * 8;
      if (full || m < n) {
        size_t idx = (size_t)m * H + ncol;
        float d = deg[m];
        d = d < 1.0f ? 1.0f : d;
        float v = out[idx] + bn + (acc_n[idx] + acc[j]) / d;
        out[idx] = v > 0.0f ? v : 0.0f;
      }
    }
  }
}

extern "C" void kernel_launch(void* const* d_in, const int* in_sizes, int n_in,
                              void* d_out, int out_size, void* d_ws, size_t ws_size,
                              hipStream_t stream) {
  const float*     x         = (const float*)d_in[0];
  const long long* ei        = (const long long*)d_in[1];
  const float*     edge_attr = (const float*)d_in[2];
  const float*     w_self_W  = (const float*)d_in[3];
  const float*     w_self_b  = (const float*)d_in[4];
  const float*     w_nei_W   = (const float*)d_in[5];
  const float*     w_edge_W  = (const float*)d_in[6];

  const int n  = in_sizes[0] / H;   // 100000 nodes
  const int ne = in_sizes[1] / 2;   // 1600000 edges
  float* out = (float*)d_out;

  // ws layout: acc_n(f32) | acc_e(f32) | deg(f32) | [64B] x_bf | x_nei_bf | wn_bf | ws_bf | we_bf
  float* base  = (float*)d_ws;
  size_t nh    = (size_t)n * H;
  float* acc_n = base;
  float* acc_e = base + nh;
  float* deg   = base + 2 * nh;
  size_t boff  = (((2 * nh + (size_t)n) * sizeof(float)) + 63) & ~(size_t)63;
  __bf16* x_bf     = (__bf16*)((char*)d_ws + boff);
  __bf16* x_nei_bf = x_bf + nh;
  __bf16* wn_bf    = x_nei_bf + nh;
  __bf16* ws_bf    = wn_bf + (size_t)H * H;
  __bf16* we_bf    = ws_bf + (size_t)H * H;

  const int gemm_blocks = (n + 127) / 128;

  // 1) zero acc_n|acc_e|deg (contiguous), convert x and the three weight matrices to bf16
  gnn_init<<<2048, 256, 0, stream>>>(acc_n, 2 * nh + (size_t)n,
                                     x, x_bf, nh,
                                     w_nei_W, w_self_W, w_edge_W,
                                     wn_bf, ws_bf, we_bf);
  // 2) node GEMMs (async LDS staging): x_nei_bf -> ws, x_self -> d_out
  gnn_node_gemm<<<gemm_blocks, 256, 0, stream>>>(x_bf, wn_bf, ws_bf, x_nei_bf, out, n);
  // 3) edge scatter: one wave per edge
  gnn_edge_scatter<<<(ne + 7) / 8, 256, 0, stream>>>(ei, edge_attr, x_nei_bf,
                                                     acc_n, acc_e, deg, ne);
  // 4) aggregated-edge GEMM + fused normalize/bias/ReLU epilogue
  gnn_edge_gemm_epilogue<<<gemm_blocks, 256, 0, stream>>>(acc_e, we_bf, acc_n, deg,
                                                          w_self_b, out, n);
}
